// DGCNN_Cline_89489938579859
// MI455X (gfx1250) — compile-verified, matrix operand order checked by reading, not run on hardware
//
#include <hip/hip_runtime.h>

#define KNN 20

typedef __bf16 bf16_t;
typedef bf16_t v16bf __attribute__((ext_vector_type(16)));
typedef float v8f __attribute__((ext_vector_type(8)));

struct Frag32 {
  uint4 lo, hi;
};  // 32 bytes -> bit_cast to v16bf

__device__ __forceinline__ float bf2f(unsigned short u) {
  unsigned int x = ((unsigned int)u) << 16;
  return __builtin_bit_cast(float, x);
}
__device__ __forceinline__ unsigned short f2bf(float f) {
  unsigned int x = __builtin_bit_cast(unsigned int, f);
  x = x + 0x7fffu + ((x >> 16) & 1u);  // round-to-nearest-even
  return (unsigned short)(x >> 16);
}

// ---------------- converts ----------------
// dst (M x Kdst) compact with zero padding beyond Ksrc; src row stride sld
__global__ void cvt_mat_bf16_k(const float* __restrict__ s, int sld,
                               unsigned short* __restrict__ d, int M, int Ksrc, int Kdst) {
  int i = blockIdx.x * blockDim.x + threadIdx.x;
  if (i >= M * Kdst) return;
  int m = i / Kdst, c = i % Kdst;
  d[i] = (c < Ksrc) ? f2bf(s[m * sld + c]) : (unsigned short)0;
}
// dst (N x Kd) row-major = transpose of src (C x N); zero pad c >= C
__global__ void cvt_t_bf16_k(const float* __restrict__ s, int C, int N,
                             unsigned short* __restrict__ d, int Kd) {
  int i = blockIdx.x * blockDim.x + threadIdx.x;
  if (i >= N * Kd) return;
  int n = i / Kd, c = i % Kd;
  d[i] = (c < C) ? f2bf(s[c * N + n]) : (unsigned short)0;
}

// ---------------- small VALU stages (STN, reductions, head tail) ----------------
__global__ void conv1d_k(const float* __restrict__ W, const float* __restrict__ sc,
                         const float* __restrict__ bi, const float* __restrict__ x, int xrs,
                         float* __restrict__ out, int ors, int O, int C, int N, int act) {
  int i = blockIdx.x * blockDim.x + threadIdx.x;
  if (i >= O * N) return;
  int o = i / N, n = i % N;
  float acc = 0.f;
  for (int c = 0; c < C; ++c) acc += W[o * C + c] * x[c * xrs + n];
  float v = sc[o] * acc + bi[o];
  if (act == 1) v = fmaxf(v, 0.f);
  else if (act == 2) v = (v >= 0.f) ? v : 0.2f * v;
  out[o * ors + n] = v;
}

__global__ void rowmax_f32_k(const float* __restrict__ x, int C, int N, int rs,
                             float* __restrict__ out) {
  int c = blockIdx.x * blockDim.x + threadIdx.x;
  if (c >= C) return;
  float m = -3.0e38f;
  for (int n = 0; n < N; ++n) m = fmaxf(m, x[c * rs + n]);
  out[c] = m;
}
// x: (Nrows x C) bf16 row-major; out[c] = max over rows (coalesced across c-threads)
__global__ void colmax_bf16_k(const unsigned short* __restrict__ x, int C, int Nrows,
                              float* __restrict__ out) {
  int c = blockIdx.x * blockDim.x + threadIdx.x;
  if (c >= C) return;
  float m = -3.0e38f;
  for (int n = 0; n < Nrows; ++n) m = fmaxf(m, bf2f(x[(size_t)n * C + c]));
  out[c] = m;
}

__global__ void fc_k(const float* __restrict__ W, const float* __restrict__ fb,
                     const float* __restrict__ sc, const float* __restrict__ bi,
                     const float* __restrict__ x, float* __restrict__ out, int O, int C,
                     int addeye, int act) {
  int o = blockIdx.x * blockDim.x + threadIdx.x;
  if (o >= O) return;
  float acc = 0.f;
  for (int c = 0; c < C; ++c) acc += W[o * C + c] * x[c];
  if (fb) acc += fb[o];
  if (sc) acc = acc * sc[o] + bi[o];
  if (act == 1) acc = fmaxf(acc, 0.f);
  if (addeye && (o % 4 == 0) && o < 9) acc += 1.f;  // eye(3) flattened: 0,4,8
  out[o] = acc;
}

__global__ void apply_tnet_k(const float* __restrict__ xin, int C, int N,
                             const float* __restrict__ t9, float* __restrict__ out) {
  int i = blockIdx.x * blockDim.x + threadIdx.x;
  if (i >= C * N) return;
  int c = i / N, n = i % N;
  float v;
  if (c < 3) {
    v = xin[0 * N + n] * t9[0 * 3 + c] + xin[1 * N + n] * t9[1 * 3 + c] +
        xin[2 * N + n] * t9[2 * 3 + c];
  } else {
    v = xin[c * N + n];
  }
  out[c * N + n] = v;
}

// ---------------- KNN (top-20 by -dist^2, self included, matches top_k) ----------------
__global__ void knn_k(const float* __restrict__ x, int C, int N, int rs, int* __restrict__ idx) {
  int n = blockIdx.x * blockDim.x + threadIdx.x;
  if (n >= N) return;
  float best[KNN];
  int bi[KNN];
  for (int t = 0; t < KNN; ++t) { best[t] = -3.0e38f; bi[t] = 0; }
  for (int m = 0; m < N; ++m) {
    float d = 0.f;
    for (int c = 0; c < C; ++c) {
      float df = x[c * rs + n] - x[c * rs + m];
      d -= df * df;
    }
    if (d > best[KNN - 1]) {
      int p = KNN - 1;
      while (p > 0 && best[p - 1] < d) { best[p] = best[p - 1]; bi[p] = bi[p - 1]; --p; }
      best[p] = d; bi[p] = m;
    }
  }
  for (int t = 0; t < KNN; ++t) idx[n * KNN + t] = bi[t];
}

// Et (N*KNN x Kp) bf16, one row per edge point: cols 0..C-1 = nbr-ctr, C..2C-1 = ctr, rest 0
__global__ void gather_edge_k(const float* __restrict__ x, int C, int N, int rs,
                              const int* __restrict__ idx, unsigned short* __restrict__ Et,
                              int Kp) {
  int nk = blockIdx.x * blockDim.x + threadIdx.x;
  if (nk >= N * KNN) return;
  int n = nk / KNN;
  int m = idx[nk];
  unsigned short* row = Et + (size_t)nk * Kp;
  for (int c = 0; c < C; ++c) {
    float ctr = x[c * rs + n];
    float nbr = x[c * rs + m];
    row[c] = f2bf(nbr - ctr);
    row[C + c] = f2bf(ctr);
  }
  for (int c = 2 * C; c < Kp; ++c) row[c] = 0;
}

// H (O x N*KNN) row-major -> out[o,n] = max over k
__global__ void maxk_k(const unsigned short* __restrict__ H, int O, int N,
                       float* __restrict__ out, int ldo) {
  int i = blockIdx.x * blockDim.x + threadIdx.x;
  if (i >= O * N) return;
  int o = i / N, n = i % N;
  float m = -3.0e38f;
  for (int j = 0; j < KNN; ++j) m = fmaxf(m, bf2f(H[o * (N * KNN) + n * KNN + j]));
  out[o * ldo + n] = m;
}

__global__ void gemv_t0_k(const float* __restrict__ hw0, const float* __restrict__ g,
                          float* __restrict__ t0, int O, int Koff, int ld) {
  int o = blockIdx.x * blockDim.x + threadIdx.x;
  if (o >= O) return;
  float acc = 0.f;
  for (int c = 0; c < 4096; ++c) acc += hw0[o * ld + Koff + c] * g[c];
  t0[o] = acc;
}

// final 512 -> 3 conv + bias; h1T is (N x 512) row-major (contiguous channel loop)
__global__ void head2_k(const float* __restrict__ W, const float* __restrict__ b3,
                        const unsigned short* __restrict__ h1T, int N, float* __restrict__ out) {
  int i = blockIdx.x * blockDim.x + threadIdx.x;
  if (i >= 3 * N) return;
  int o = i / N, n = i % N;
  float acc = 0.f;
  for (int c = 0; c < 512; ++c) acc += W[o * 512 + c] * bf2f(h1T[(size_t)n * 512 + c]);
  out[o * N + n] = acc + b3[o];
}

// ---------------- WMMA bf16 GEMM, fused affine+activation epilogue ----------------
// D[M,N] = act(scale[m]*(A x Bt^T + extra[m]) + bias[m])
// A: (M x K) row-major, lda%8==0.  Bt: (N x K) row-major (i.e. B transposed), ldb%8==0.
// Preconditions: M%64==0, N%16==0, K%32==0. One wave = 64x16 strip (4 WMMA tiles, shared B).
// Both fragments are two contiguous 16B runs per lane -> all b128 loads, no branches.
// outT=0: store D[m*ldo+n]; outT=1: store transposed D[n*ldo+m] (feeds next GEMM's Bt).
__global__ __launch_bounds__(256) void gemm_bf16_k(
    const unsigned short* __restrict__ A, int lda, const unsigned short* __restrict__ Bt, int ldb,
    unsigned short* __restrict__ outH, int ldo, int outT, const float* __restrict__ scale,
    const float* __restrict__ bias, const float* __restrict__ extra, int M, int N, int K,
    int act) {
  int wave = blockIdx.x * (blockDim.x >> 5) + (threadIdx.x >> 5);
  int lane = threadIdx.x & 31;
  int ntn = N >> 4;
  int nstrip = M >> 6;
  if (wave >= ntn * nstrip) return;
  int ms = (wave / ntn) << 6;
  int col0 = (wave % ntn) << 4;
  int hs = lane >> 4;  // ISA 7.12.2: lanes 16-31 carry the K+8 element group
  int l15 = lane & 15;
  const unsigned short* Arow = A + (ms + l15) * lda + (hs << 3);   // + t*16*lda + k0
  const unsigned short* Brow = Bt + (col0 + l15) * ldb + (hs << 3);  // + k0
  v8f acc[4];
#pragma unroll
  for (int t = 0; t < 4; ++t) acc[t] = (v8f){0.f, 0.f, 0.f, 0.f, 0.f, 0.f, 0.f, 0.f};

  for (int k0 = 0; k0 < K; k0 += 32) {
    const uint4* bp = (const uint4*)(Brow + k0);
    Frag32 bf;
    bf.lo = bp[0];  // k0+hs*8    .. +7
    bf.hi = bp[2];  // k0+16+hs*8 .. +7
    v16bf bfrag = __builtin_bit_cast(v16bf, bf);
    if (k0 + 32 < K) {  // global_prefetch_b8 next K block of both streams
      __builtin_prefetch((const void*)(Brow + k0 + 32), 0, 0);
      __builtin_prefetch((const void*)(Arow + k0 + 32), 0, 0);
    }
#pragma unroll
    for (int t = 0; t < 4; ++t) {
      const uint4* ap = (const uint4*)(Arow + t * 16 * lda + k0);
      Frag32 af;
      af.lo = ap[0];
      af.hi = ap[2];
      v16bf a = __builtin_bit_cast(v16bf, af);
      acc[t] = __builtin_amdgcn_wmma_f32_16x16x32_bf16(false, a, false, bfrag, (short)0, acc[t],
                                                       false, false);
    }
  }

  int nn = col0 + l15;
#pragma unroll
  for (int t = 0; t < 4; ++t) {
#pragma unroll
    for (int r = 0; r < 8; ++r) {
      int mm = ms + (t << 4) + r + (hs << 3);  // D: VGPR r, lanes>=16 -> M=r+8
      float v = acc[t][r];
      if (extra) v += extra[mm];
      v = v * scale[mm] + bias[mm];
      if (act == 1) v = fmaxf(v, 0.f);
      else if (act == 2) v = (v >= 0.f) ? v : 0.2f * v;
      if (outT) outH[(size_t)nn * ldo + mm] = f2bf(v);
      else outH[(size_t)mm * ldo + nn] = f2bf(v);
    }
  }
}

#define LAUNCH(kern, n, bs, ...) kern<<<((n) + (bs)-1) / (bs), (bs), 0, stream>>>(__VA_ARGS__)

static inline int kpad32(int k) { return (k + 31) & ~31; }

extern "C" void kernel_launch(void* const* d_in, const int* in_sizes, int n_in, void* d_out,
                              int out_size, void* d_ws, size_t ws_size, hipStream_t stream) {
  (void)in_sizes; (void)out_size; (void)ws_size;
  const int B = 2, Nobs = 2048, Nbnd = 1024;
  const float* F[64];
  for (int i = 0; i < n_in && i < 64; ++i) F[i] = (const float*)d_in[i];

  const float* observation = F[0];
  const float* boundary = F[1];
  const float *w1 = F[2], *s1 = F[3], *b1 = F[4], *w2 = F[5], *s2 = F[6], *b2 = F[7];
  const float *w3 = F[8], *s3 = F[9], *b3 = F[10];
  const float *fw1 = F[11], *fb1 = F[12], *s4 = F[13], *b4 = F[14];
  const float *fw2 = F[15], *fb2 = F[16], *s5 = F[17], *b5 = F[18];
  const float *fw3 = F[19], *fb3 = F[20];

  // edge layers: obs (4->64, 64->128, 128->256), bnd (5->64, 64->64, 64->128)
  const int eC[6] = {4, 64, 128, 5, 64, 64};
  const int eO[6] = {64, 128, 256, 64, 64, 128};
  const int eW[6] = {21, 24, 27, 30, 33, 36};

  char* base = (char*)d_ws;
  size_t off = 0;
  auto alloc = [&](size_t bytes) -> char* {
    char* p = base + off;
    off = (off + bytes + 255) & ~(size_t)255;
    return p;
  };

  unsigned short* wE[6];
  for (int l = 0; l < 6; ++l) wE[l] = (unsigned short*)alloc((size_t)eO[l] * kpad32(2 * eC[l]) * 2);
  unsigned short* wProjO = (unsigned short*)alloc((size_t)4096 * 448 * 2);
  unsigned short* wProjB = (unsigned short*)alloc((size_t)4096 * 256 * 2);
  unsigned short* wH0 = (unsigned short*)alloc((size_t)1024 * 4096 * 2);
  unsigned short* wH1 = (unsigned short*)alloc((size_t)512 * 1024 * 2);
  float* aObs = (float*)alloc((size_t)B * 4 * Nobs * 4);
  float* aBnd = (float*)alloc((size_t)B * 5 * Nbnd * 4);
  float* sh1 = (float*)alloc((size_t)64 * Nobs * 4);
  float* sh2 = (float*)alloc((size_t)128 * Nobs * 4);
  float* sh3 = (float*)alloc((size_t)1024 * Nobs * 4);
  float* sg = (float*)alloc(1024 * 4);
  float* sf1 = (float*)alloc(512 * 4);
  float* sf2 = (float*)alloc(256 * 4);
  float* st = (float*)alloc(16 * 4);
  float* featsO = (float*)alloc((size_t)B * 448 * Nobs * 4);
  float* featsB = (float*)alloc((size_t)B * 256 * Nbnd * 4);
  unsigned short* featsOT = (unsigned short*)alloc((size_t)B * Nobs * 448 * 2);  // (N x 448)
  unsigned short* featsBT = (unsigned short*)alloc((size_t)B * Nbnd * 256 * 2);  // (N x 256)
  int* idxb = (int*)alloc((size_t)Nobs * KNN * 4);
  float* bndg = (float*)alloc((size_t)B * 4096 * 4);
  float* t0 = (float*)alloc((size_t)B * 1024 * 4);
  // phase-overlapped union region (edge scratch, then GEMM activations)
  char* U = alloc(62914560);
  unsigned short* Et = (unsigned short*)U;                 // up to 40960 x 512 bf16
  unsigned short* Hbuf = (unsigned short*)(U + 41943040);  // up to 256 x 40960 bf16
  unsigned short* obsFT = (unsigned short*)U;              // B x (2048 x 4096) bf16
  unsigned short* bndFT = (unsigned short*)(U + 33554432); // B x (1024 x 4096) bf16
  unsigned short* h0T = (unsigned short*)(U + 50331648);   // B x (2048 x 1024) bf16
  unsigned short* h1T = (unsigned short*)(U + 58720256);   // B x (2048 x 512) bf16

  // ---- bf16 weight staging (K padded to multiples of 32) ----
  for (int l = 0; l < 6; ++l) {
    int ks = 2 * eC[l], kd = kpad32(ks);
    LAUNCH(cvt_mat_bf16_k, eO[l] * kd, 256, F[eW[l]], ks, wE[l], eO[l], ks, kd);
  }
  LAUNCH(cvt_mat_bf16_k, 4096 * 448, 256, F[39], 448, wProjO, 4096, 448, 448);
  LAUNCH(cvt_mat_bf16_k, 4096 * 256, 256, F[42], 256, wProjB, 4096, 256, 256);
  LAUNCH(cvt_mat_bf16_k, 1024 * 4096, 256, F[45], 8192, wH0, 1024, 4096, 4096);
  LAUNCH(cvt_mat_bf16_k, 512 * 1024, 256, F[48], 1024, wH1, 512, 1024, 1024);

  // ---- STN + alignment for both clouds ----
  for (int cl = 0; cl < 2; ++cl) {
    const float* xcl = (cl == 0) ? observation : boundary;
    int C = (cl == 0) ? 4 : 5;
    int N = (cl == 0) ? Nobs : Nbnd;
    float* al = (cl == 0) ? aObs : aBnd;
    for (int b = 0; b < B; ++b) {
      const float* xb = xcl + (size_t)b * C * N;
      LAUNCH(conv1d_k, 64 * N, 256, w1, s1, b1, xb, N, sh1, N, 64, 3, N, 1);
      LAUNCH(conv1d_k, 128 * N, 256, w2, s2, b2, sh1, N, sh2, N, 128, 64, N, 1);
      LAUNCH(conv1d_k, 1024 * N, 256, w3, s3, b3, sh2, N, sh3, N, 1024, 128, N, 1);
      LAUNCH(rowmax_f32_k, 1024, 256, sh3, 1024, N, N, sg);
      LAUNCH(fc_k, 512, 128, fw1, fb1, s4, b4, sg, sf1, 512, 1024, 0, 1);
      LAUNCH(fc_k, 256, 128, fw2, fb2, s5, b5, sf1, sf2, 256, 512, 0, 1);
      LAUNCH(fc_k, 9, 32, fw3, fb3, (const float*)nullptr, (const float*)nullptr, sf2, st, 9, 256,
             1, 0);
      LAUNCH(apply_tnet_k, C * N, 256, xb, C, N, st, al + (size_t)b * C * N);
    }
  }

  // ---- edge-conv stacks: KNN -> gather Et -> WMMA GEMM (affine+lrelu) -> max over k ----
  for (int cl = 0; cl < 2; ++cl) {
    int N = (cl == 0) ? Nobs : Nbnd;
    int Ctot = (cl == 0) ? 448 : 256;
    float* feats = (cl == 0) ? featsO : featsB;
    float* al = (cl == 0) ? aObs : aBnd;
    int C0 = (cl == 0) ? 4 : 5;
    int NK = N * KNN;
    for (int b = 0; b < B; ++b) {
      const float* cur = al + (size_t)b * C0 * N;
      int C = C0, offc = 0;
      for (int l = 0; l < 3; ++l) {
        int li = cl * 3 + l;
        int O = eO[li];
        int Kp = kpad32(2 * C);
        LAUNCH(knn_k, N, 128, cur, C, N, N, idxb);
        LAUNCH(gather_edge_k, NK, 256, cur, C, N, N, idxb, Et, Kp);
        int waves = (O >> 6) * (NK >> 4);
        gemm_bf16_k<<<(waves + 7) / 8, 256, 0, stream>>>(wE[li], Kp, Et, Kp, Hbuf, NK, 0,
                                                         F[eW[li] + 1], F[eW[li] + 2], nullptr, O,
                                                         NK, Kp, 2);
        float* outc = feats + (size_t)b * Ctot * N + (size_t)offc * N;
        LAUNCH(maxk_k, O * N, 256, Hbuf, O, N, outc, N);
        cur = outc;
        C = O;
        offc += O;
      }
    }
  }

  // ---- projections, boundary pooling, head (all B-inputs N-major) ----
  for (int b = 0; b < B; ++b) {
    LAUNCH(cvt_t_bf16_k, Nobs * 448, 256, featsO + (size_t)b * 448 * Nobs, 448, Nobs,
           featsOT + (size_t)b * Nobs * 448, 448);
    LAUNCH(cvt_t_bf16_k, Nbnd * 256, 256, featsB + (size_t)b * 256 * Nbnd, 256, Nbnd,
           featsBT + (size_t)b * Nbnd * 256, 256);
  }

  for (int b = 0; b < B; ++b) {
    {  // obs proj: 448 -> 4096, lrelu(bn); output transposed (N x 4096) for head0
      int waves = (4096 >> 6) * (Nobs >> 4);
      gemm_bf16_k<<<(waves + 7) / 8, 256, 0, stream>>>(
          wProjO, 448, featsOT + (size_t)b * Nobs * 448, 448, obsFT + (size_t)b * Nobs * 4096,
          4096, 1, F[40], F[41], nullptr, 4096, Nobs, 448, 2);
    }
    {  // bnd proj: 256 -> 4096; output transposed (N x 4096) for column-max pooling
      int waves = (4096 >> 6) * (Nbnd >> 4);
      gemm_bf16_k<<<(waves + 7) / 8, 256, 0, stream>>>(
          wProjB, 256, featsBT + (size_t)b * Nbnd * 256, 256, bndFT + (size_t)b * Nbnd * 4096,
          4096, 1, F[43], F[44], nullptr, 4096, Nbnd, 256, 2);
    }
    LAUNCH(colmax_bf16_k, 4096, 256, bndFT + (size_t)b * Nbnd * 4096, 4096, Nbnd, bndg + b * 4096);
    LAUNCH(gemv_t0_k, 1024, 128, F[45], bndg + b * 4096, t0 + b * 1024, 1024, 4096, 8192);
    {  // head0: obs half via WMMA (B = obsFT), boundary half as per-row extra; out (N x 1024)
      int waves = (1024 >> 6) * (Nobs >> 4);
      gemm_bf16_k<<<(waves + 7) / 8, 256, 0, stream>>>(
          wH0, 4096, obsFT + (size_t)b * Nobs * 4096, 4096, h0T + (size_t)b * Nobs * 1024, 1024,
          1, F[46], F[47], t0 + b * 1024, 1024, Nobs, 4096, 2);
    }
    {  // head1: 1024 -> 512; out (N x 512)
      int waves = (512 >> 6) * (Nobs >> 4);
      gemm_bf16_k<<<(waves + 7) / 8, 256, 0, stream>>>(
          wH1, 1024, h0T + (size_t)b * Nobs * 1024, 1024, h1T + (size_t)b * Nobs * 512, 512, 1,
          F[49], F[50], nullptr, 512, Nobs, 1024, 2);
    }
    LAUNCH(head2_k, 3 * Nobs, 256, F[51], F[52], h1T + (size_t)b * Nobs * 512, Nobs,
           (float*)d_out + (size_t)b * 3 * Nobs);
  }
}
